// ProteinGNNEncoder_27298812133695
// MI455X (gfx1250) — matrix operand order repaired and checked by reading, not compiled, not run on hardware
//
#include <hip/hip_runtime.h>

#define N_NODES 10000
#define N_EDGES 160000
#define EP_EDGES 170000   // N_EDGES + N_NODES self loops
#define IN_F 20
#define HIDc 256
#define LATc 1024

typedef __bf16 bf16_t;
typedef __attribute__((ext_vector_type(16))) __bf16 v16bf;
typedef __attribute__((ext_vector_type(8)))  float v8f;
typedef __attribute__((ext_vector_type(4)))  float f32x4;

union bfrag { v16bf v; f32x4 q[2]; };

// ---------------------------------------------------------------------------
// WMMA GEMM: C[M,N] = A[M,K](bf16, row major) * Bpacked[K,N](bf16, fragment
// packed).  One wave per 16x16 output tile, K in steps of 32 via
// v_wmma_f32_16x16x32_bf16 with f32 accumulation.
// A fragment (16-bit 16x32): lanes 0-15 row=lane, K=[0..7]+[16..23];
//                            lanes 16-31 row=lane-16, K=[8..15]+[24..31].
// B packed layout: per (ntile,ktile) tile, lane-major 16 contiguous bf16:
//   lanes 0-15: col=lane,    K=kt*32+[0..7]+[16..23]
//   lanes 16-31: col=lane-16, K=kt*32+[8..15]+[24..31]
// ---------------------------------------------------------------------------
__global__ void __launch_bounds__(128)
wmma_gemm_bf16(const bf16_t* __restrict__ A, const bf16_t* __restrict__ Bp,
               float* __restrict__ C, int Mtiles, int Ntiles, int Ktiles,
               int lda, int ldc) {
  const int wave = blockIdx.x * (blockDim.x >> 5) + (threadIdx.x >> 5);
  if (wave >= Mtiles * Ntiles) return;          // whole-wave guard: EXEC all-1s
  const int lane = threadIdx.x & 31;
  const int nt = wave % Ntiles;
  const int mt = wave / Ntiles;
  const int half = lane >> 4;
  const int l15 = lane & 15;
  const int rowA = mt * 16 + l15;
  const int kloA = half * 8;
  const bf16_t* arow = A + (size_t)rowA * lda + kloA;
  const bf16_t* bbase = Bp + ((size_t)nt * Ktiles * 32 + lane) * 16;
  v8f acc = {};
  for (int kt = 0; kt < Ktiles; ++kt) {
    bfrag a, b;
    a.q[0] = *(const f32x4*)(arow + kt * 32);        // K = klo..klo+7
    a.q[1] = *(const f32x4*)(arow + kt * 32 + 16);   // K = klo+16..klo+23
    const f32x4* bp = (const f32x4*)(bbase + (size_t)kt * 512);
    b.q[0] = bp[0];
    b.q[1] = bp[1];
    acc = __builtin_amdgcn_wmma_f32_16x16x32_bf16(false, a.v, false, b.v,
                                                  (short)0, acc, false, false);
  }
  // f32 C/D layout: VGPR r -> row = r + 8*(lane>=16), col = lane&15
  const int col = nt * 16 + l15;
  const int rbase = mt * 16 + half * 8;
#pragma unroll
  for (int r = 0; r < 8; ++r)
    C[(size_t)(rbase + r) * ldc + col] = acc[r];
}

// ---------------------------------------------------------------------------
__global__ void fill_kernel(float* p, float v, int n) {
  int i = blockIdx.x * blockDim.x + threadIdx.x;
  if (i < n) p[i] = v;
}

__global__ void deg_kernel(const int* __restrict__ dst,
                           const float* __restrict__ ea,
                           float* __restrict__ cnt, float* __restrict__ sumea) {
  int e = blockIdx.x * blockDim.x + threadIdx.x;
  if (e >= N_EDGES) return;
  int d = dst[e];
  atomicAdd(&cnt[d], 1.0f);
  atomicAdd(&sumea[d * 3 + 0], ea[e * 3 + 0]);
  atomicAdd(&sumea[d * 3 + 1], ea[e * 3 + 1]);
  atomicAdd(&sumea[d * 3 + 2], ea[e * 3 + 2]);
}

__global__ void pack_x_kernel(const float* __restrict__ x, bf16_t* __restrict__ xbf) {
  int idx = blockIdx.x * blockDim.x + threadIdx.x;
  if (idx >= N_NODES * 32) return;
  int n = idx >> 5, k = idx & 31;
  xbf[idx] = (k < IN_F) ? (bf16_t)x[n * IN_F + k] : (bf16_t)0.0f;
}

__global__ void pack_b_kernel(const float* __restrict__ W, bf16_t* __restrict__ Bp,
                              int K, int Nw, int Ktiles, int Ntiles) {
  int idx = blockIdx.x * blockDim.x + threadIdx.x;
  if (idx >= Ntiles * Ktiles * 512) return;
  int j = idx & 15;
  int lane = (idx >> 4) & 31;
  int t = idx >> 9;
  int kt = t % Ktiles, nt = t / Ktiles;
  int nn = nt * 16 + (lane & 15);
  int klo = (lane < 16) ? 0 : 8;
  int k = kt * 32 + klo + ((j < 8) ? j : j + 8);
  Bp[idx] = (k < K) ? (bf16_t)W[(size_t)k * Nw + nn] : (bf16_t)0.0f;
}

// als/ald per node: dot of each head slice with a_src/a_dst
__global__ void __launch_bounds__(256)
attn_node_kernel(const float* __restrict__ hmat, const float* __restrict__ asrc,
                 const float* __restrict__ adst, float* __restrict__ als,
                 float* __restrict__ ald, int H) {
  __shared__ float red[256];
  const int n = blockIdx.x, c = threadIdx.x;
  for (int h = 0; h < H; ++h) {
    float v = hmat[(size_t)n * H * 256 + h * 256 + c];
    red[c] = v * asrc[h * 256 + c];
    __syncthreads();
    for (int s = 128; s > 0; s >>= 1) { if (c < s) red[c] += red[c + s]; __syncthreads(); }
    if (c == 0) als[n * H + h] = red[0];
    __syncthreads();
    red[c] = v * adst[h * 256 + c];
    __syncthreads();
    for (int s = 128; s > 0; s >>= 1) { if (c < s) red[c] += red[c + s]; __syncthreads(); }
    if (c == 0) ald[n * H + h] = red[0];
    __syncthreads();
  }
}

// M[j,h] = sum_c We[j, h*256+c] * a_e[h,c]  (3 x H, collapses the E x 1024 GEMM)
__global__ void mmat_kernel(const float* __restrict__ We, const float* __restrict__ ae,
                            float* __restrict__ M, int H) {
  int t = threadIdx.x;
  if (t >= 3 * H) return;
  int j = t / H, h = t % H;
  float s = 0.f;
  for (int c = 0; c < 256; ++c) s += We[(size_t)j * H * 256 + h * 256 + c] * ae[h * 256 + c];
  M[j * H + h] = s;
}

__global__ void edge_logit_kernel(const int* __restrict__ src, const int* __restrict__ dst,
                                  const float* __restrict__ ea, const float* __restrict__ sumea,
                                  const float* __restrict__ cnt, const float* __restrict__ als,
                                  const float* __restrict__ ald, const float* __restrict__ M,
                                  float* __restrict__ alpha, int H) {
  int e = blockIdx.x * blockDim.x + threadIdx.x;
  if (e >= EP_EDGES) return;
  int s, d; float e0, e1, e2;
  if (e < N_EDGES) {
    s = src[e]; d = dst[e];
    e0 = ea[e * 3 + 0]; e1 = ea[e * 3 + 1]; e2 = ea[e * 3 + 2];
  } else {  // self loop, edge attr = mean of incoming
    s = d = e - N_EDGES;
    float inv = 1.0f / fmaxf(cnt[d], 1.0f);
    e0 = sumea[d * 3 + 0] * inv; e1 = sumea[d * 3 + 1] * inv; e2 = sumea[d * 3 + 2] * inv;
  }
  for (int h = 0; h < H; ++h) {
    float ale = e0 * M[0 * H + h] + e1 * M[1 * H + h] + e2 * M[2 * H + h];
    float z = als[s * H + h] + ald[d * H + h] + ale;
    alpha[(size_t)e * H + h] = (z > 0.f) ? z : 0.2f * z;   // leaky_relu(0.2)
  }
}

__device__ __forceinline__ void atomicMaxF(float* addr, float val) {
  if (val >= 0.f) atomicMax((int*)addr, __float_as_int(val));
  else            atomicMin((unsigned int*)addr, __float_as_uint(val));
}

__global__ void seg_max_kernel(const int* __restrict__ dst, const float* __restrict__ alpha,
                               float* __restrict__ m, int H) {
  int e = blockIdx.x * blockDim.x + threadIdx.x;
  if (e >= EP_EDGES) return;
  int d = (e < N_EDGES) ? dst[e] : (e - N_EDGES);
  for (int h = 0; h < H; ++h) atomicMaxF(&m[d * H + h], alpha[(size_t)e * H + h]);
}

__global__ void exp_sum_kernel(const int* __restrict__ dst, float* __restrict__ alpha,
                               const float* __restrict__ m, float* __restrict__ ssum, int H) {
  int e = blockIdx.x * blockDim.x + threadIdx.x;
  if (e >= EP_EDGES) return;
  int d = (e < N_EDGES) ? dst[e] : (e - N_EDGES);
  for (int h = 0; h < H; ++h) {
    float ex = __expf(alpha[(size_t)e * H + h] - m[d * H + h]);
    alpha[(size_t)e * H + h] = ex;
    atomicAdd(&ssum[d * H + h], ex);
  }
}

// out[d, :] += (ex/sum) * h[s, :]  — h matrix fits in the 192MB L2, so the
// gather and the f32 scatter-atomics both resolve in-cache.
__global__ void __launch_bounds__(256)
aggregate_kernel(const int* __restrict__ src, const int* __restrict__ dst,
                 const float* __restrict__ alpha, const float* __restrict__ ssum,
                 const float* __restrict__ hmat, float* __restrict__ out, int H) {
  const int e = blockIdx.x;
  const int s = (e < N_EDGES) ? src[e] : (e - N_EDGES);
  const int d = (e < N_EDGES) ? dst[e] : (e - N_EDGES);
  const int HC = H * 256;
  for (int col = threadIdx.x; col < HC; col += 256) {
    int h = col >> 8;
    float w = alpha[(size_t)e * H + h] / (ssum[d * H + h] + 1e-16f);
    atomicAdd(&out[(size_t)d * HC + col], w * hmat[(size_t)s * HC + col]);
  }
}

// bias + ELU + LayerNorm; optionally emits bf16 copy (A-matrix of next GEMM)
__global__ void __launch_bounds__(256)
epilogue_kernel(const float* __restrict__ acc, const float* __restrict__ bias,
                const float* __restrict__ g, const float* __restrict__ be,
                float* __restrict__ outf, bf16_t* __restrict__ outbf, int D) {
  __shared__ float red[256];
  const int n = blockIdx.x, t = threadIdx.x;
  const int per = D >> 8;
  float vloc[4];
  float sum = 0.f, sq = 0.f;
  for (int i = 0; i < per; ++i) {
    int col = t + i * 256;
    float v = acc[(size_t)n * D + col] + bias[col];
    v = (v > 0.f) ? v : (__expf(v) - 1.0f);     // ELU
    vloc[i] = v; sum += v; sq += v * v;
  }
  red[t] = sum; __syncthreads();
  for (int s = 128; s > 0; s >>= 1) { if (t < s) red[t] += red[t + s]; __syncthreads(); }
  float mean = red[0] / (float)D; __syncthreads();
  red[t] = sq; __syncthreads();
  for (int s = 128; s > 0; s >>= 1) { if (t < s) red[t] += red[t + s]; __syncthreads(); }
  float var = red[0] / (float)D - mean * mean;
  float rstd = rsqrtf(var + 1e-5f);
  for (int i = 0; i < per; ++i) {
    int col = t + i * 256;
    float y = g[col] * (vloc[i] - mean) * rstd + be[col];
    if (outf)  outf[(size_t)n * D + col] = y;
    if (outbf) outbf[(size_t)n * D + col] = (bf16_t)y;
  }
}

__global__ void __launch_bounds__(256)
pool_kernel(const float* __restrict__ h, float* __restrict__ pooled) {
  __shared__ float rs[256], rm[256];
  const int dco = blockIdx.x, t = threadIdx.x;
  float sum = 0.f, mx = -__builtin_inff();
  for (int n = t; n < N_NODES; n += 256) {
    float v = h[(size_t)n * 256 + dco];
    sum += v; mx = fmaxf(mx, v);
  }
  rs[t] = sum; rm[t] = mx; __syncthreads();
  for (int s = 128; s > 0; s >>= 1) {
    if (t < s) { rs[t] += rs[t + s]; rm[t] = fmaxf(rm[t], rm[t + s]); }
    __syncthreads();
  }
  if (t == 0) { pooled[dco] = rs[0] / (float)N_NODES; pooled[256 + dco] = rm[0]; }
}

__global__ void __launch_bounds__(256)
fc_kernel(const float* __restrict__ pooled, const float* __restrict__ W,
          const float* __restrict__ b, float* __restrict__ out) {
  int j = blockIdx.x * 256 + threadIdx.x;   // grid = 4
  float acc = b[j];
  for (int i = 0; i < 512; ++i) acc += pooled[i] * W[(size_t)i * LATc + j];
  out[j] = fmaxf(acc, 0.f);                 // ReLU
}

__global__ void __launch_bounds__(256)
final_ln_kernel(const float* __restrict__ fc, const float* __restrict__ g,
                const float* __restrict__ b, float* __restrict__ out) {
  __shared__ float red[256];
  const int t = threadIdx.x;
  float vloc[4], sum = 0.f, sq = 0.f;
  for (int i = 0; i < 4; ++i) {
    float v = fc[t + i * 256];
    vloc[i] = v; sum += v; sq += v * v;
  }
  red[t] = sum; __syncthreads();
  for (int s = 128; s > 0; s >>= 1) { if (t < s) red[t] += red[t + s]; __syncthreads(); }
  float mean = red[0] / (float)LATc; __syncthreads();
  red[t] = sq; __syncthreads();
  for (int s = 128; s > 0; s >>= 1) { if (t < s) red[t] += red[t + s]; __syncthreads(); }
  float var = red[0] / (float)LATc - mean * mean;
  float rstd = rsqrtf(var + 1e-5f);
  for (int i = 0; i < 4; ++i) {
    int col = t + i * 256;
    out[col] = g[col] * (vloc[i] - mean) * rstd + b[col];
  }
}

// ---------------------------------------------------------------------------
extern "C" void kernel_launch(void* const* d_in, const int* in_sizes, int n_in,
                              void* d_out, int out_size, void* d_ws, size_t ws_size,
                              hipStream_t stream) {
  (void)in_sizes; (void)n_in; (void)out_size; (void)ws_size;
  const float* x      = (const float*)d_in[0];
  const float* ea     = (const float*)d_in[1];
  const float* W1     = (const float*)d_in[2];
  const float* a_src1 = (const float*)d_in[3];
  const float* a_dst1 = (const float*)d_in[4];
  const float* We1    = (const float*)d_in[5];
  const float* a_e1   = (const float*)d_in[6];
  const float* b1     = (const float*)d_in[7];
  const float* g1     = (const float*)d_in[8];
  const float* be1    = (const float*)d_in[9];
  const float* W2     = (const float*)d_in[10];
  const float* a_src2 = (const float*)d_in[11];
  const float* a_dst2 = (const float*)d_in[12];
  const float* We2    = (const float*)d_in[13];
  const float* a_e2   = (const float*)d_in[14];
  const float* b2     = (const float*)d_in[15];
  const float* g2     = (const float*)d_in[16];
  const float* be2    = (const float*)d_in[17];
  const float* Wfc    = (const float*)d_in[18];
  const float* bfc    = (const float*)d_in[19];
  const float* gf     = (const float*)d_in[20];
  const float* bfv    = (const float*)d_in[21];
  const int*   ei     = (const int*)d_in[22];
  const int* src = ei;
  const int* dst = ei + N_EDGES;
  float* out = (float*)d_out;

  char* ws = (char*)d_ws;
  size_t off = 0;
  auto alloc = [&](size_t bytes) {
    size_t o = off; off += (bytes + 255) & ~(size_t)255; return o;
  };
  float*  cnt    = (float*)(ws + alloc((size_t)N_NODES * 4));
  float*  sumea  = (float*)(ws + alloc((size_t)N_NODES * 3 * 4));
  float*  M1     = (float*)(ws + alloc(64));
  float*  M2     = (float*)(ws + alloc(64));
  bf16_t* xbf    = (bf16_t*)(ws + alloc((size_t)N_NODES * 32 * 2));
  bf16_t* w1p    = (bf16_t*)(ws + alloc((size_t)64 * 512 * 2));
  bf16_t* w2p    = (bf16_t*)(ws + alloc((size_t)16 * 32 * 512 * 2));
  float*  h1     = (float*)(ws + alloc((size_t)N_NODES * 1024 * 4));
  float*  out1   = (float*)(ws + alloc((size_t)N_NODES * 1024 * 4));
  bf16_t* h1bf   = (bf16_t*)(ws + alloc((size_t)N_NODES * 1024 * 2));
  float*  als1   = (float*)(ws + alloc((size_t)N_NODES * 4 * 4));
  float*  ald1   = (float*)(ws + alloc((size_t)N_NODES * 4 * 4));
  float*  m1     = (float*)(ws + alloc((size_t)N_NODES * 4 * 4));
  float*  ssum1  = (float*)(ws + alloc((size_t)N_NODES * 4 * 4));
  float*  alpha1 = (float*)(ws + alloc((size_t)EP_EDGES * 4 * 4));
  float*  h2     = (float*)(ws + alloc((size_t)N_NODES * 256 * 4));
  float*  out2   = (float*)(ws + alloc((size_t)N_NODES * 256 * 4));
  float*  h2ln   = (float*)(ws + alloc((size_t)N_NODES * 256 * 4));
  float*  als2   = (float*)(ws + alloc((size_t)N_NODES * 4));
  float*  ald2   = (float*)(ws + alloc((size_t)N_NODES * 4));
  float*  m2     = (float*)(ws + alloc((size_t)N_NODES * 4));
  float*  ssum2  = (float*)(ws + alloc((size_t)N_NODES * 4));
  float*  alpha2 = (float*)(ws + alloc((size_t)EP_EDGES * 4));
  float*  pooled = (float*)(ws + alloc(512 * 4));
  float*  fcbuf  = (float*)(ws + alloc(1024 * 4));

  hipMemsetAsync(cnt,   0, (size_t)N_NODES * 4, stream);
  hipMemsetAsync(sumea, 0, (size_t)N_NODES * 3 * 4, stream);
  hipMemsetAsync(out1,  0, (size_t)N_NODES * 1024 * 4, stream);
  hipMemsetAsync(ssum1, 0, (size_t)N_NODES * 4 * 4, stream);
  hipMemsetAsync(out2,  0, (size_t)N_NODES * 256 * 4, stream);
  hipMemsetAsync(ssum2, 0, (size_t)N_NODES * 4, stream);
  fill_kernel<<<(N_NODES * 4 + 255) / 256, 256, 0, stream>>>(m1, -__builtin_inff(), N_NODES * 4);
  fill_kernel<<<(N_NODES + 255) / 256, 256, 0, stream>>>(m2, -__builtin_inff(), N_NODES);

  deg_kernel<<<(N_EDGES + 255) / 256, 256, 0, stream>>>(dst, ea, cnt, sumea);
  pack_x_kernel<<<(N_NODES * 32 + 255) / 256, 256, 0, stream>>>(x, xbf);
  pack_b_kernel<<<(64 * 512 + 255) / 256, 256, 0, stream>>>(W1, w1p, IN_F, 1024, 1, 64);
  pack_b_kernel<<<(16 * 32 * 512 + 255) / 256, 256, 0, stream>>>(W2, w2p, 1024, 256, 32, 16);

  // Layer 1: h1 = x @ W1  (625 x 64 tiles, K padded 20->32)
  wmma_gemm_bf16<<<(625 * 64 + 3) / 4, 128, 0, stream>>>(xbf, w1p, h1, 625, 64, 1, 32, 1024);
  attn_node_kernel<<<N_NODES, 256, 0, stream>>>(h1, a_src1, a_dst1, als1, ald1, 4);
  mmat_kernel<<<1, 32, 0, stream>>>(We1, a_e1, M1, 4);
  edge_logit_kernel<<<(EP_EDGES + 255) / 256, 256, 0, stream>>>(src, dst, ea, sumea, cnt,
                                                                als1, ald1, M1, alpha1, 4);
  seg_max_kernel<<<(EP_EDGES + 255) / 256, 256, 0, stream>>>(dst, alpha1, m1, 4);
  exp_sum_kernel<<<(EP_EDGES + 255) / 256, 256, 0, stream>>>(dst, alpha1, m1, ssum1, 4);
  aggregate_kernel<<<EP_EDGES, 256, 0, stream>>>(src, dst, alpha1, ssum1, h1, out1, 4);
  epilogue_kernel<<<N_NODES, 256, 0, stream>>>(out1, b1, g1, be1, (float*)nullptr, h1bf, 1024);

  // Layer 2: h2 = h1ln @ W2  (625 x 16 tiles, K=1024)
  wmma_gemm_bf16<<<(625 * 16 + 3) / 4, 128, 0, stream>>>(h1bf, w2p, h2, 625, 16, 32, 1024, 256);
  attn_node_kernel<<<N_NODES, 256, 0, stream>>>(h2, a_src2, a_dst2, als2, ald2, 1);
  mmat_kernel<<<1, 32, 0, stream>>>(We2, a_e2, M2, 1);
  edge_logit_kernel<<<(EP_EDGES + 255) / 256, 256, 0, stream>>>(src, dst, ea, sumea, cnt,
                                                                als2, ald2, M2, alpha2, 1);
  seg_max_kernel<<<(EP_EDGES + 255) / 256, 256, 0, stream>>>(dst, alpha2, m2, 1);
  exp_sum_kernel<<<(EP_EDGES + 255) / 256, 256, 0, stream>>>(dst, alpha2, m2, ssum2, 1);
  aggregate_kernel<<<EP_EDGES, 256, 0, stream>>>(src, dst, alpha2, ssum2, h2, out2, 1);
  epilogue_kernel<<<N_NODES, 256, 0, stream>>>(out2, b2, g2, be2, h2ln, (bf16_t*)nullptr, 256);

  // Readout: mean/max pool -> FC -> ReLU -> LN
  pool_kernel<<<256, 256, 0, stream>>>(h2ln, pooled);
  fc_kernel<<<4, 256, 0, stream>>>(pooled, Wfc, bfc, fcbuf);
  final_ln_kernel<<<1, 256, 0, stream>>>(fcbuf, gf, bfv, out);
}
